// LocalWindowAttention_61864708931989
// MI455X (gfx1250) — matrix operand order repaired and checked by reading, not compiled, not run on hardware
//
#include <hip/hip_runtime.h>
#include <math.h>
#include <cstdint>

// ---------------------------------------------------------------------------
// Problem constants (B=2, S=4096, D=1024, H=16, hd=64, W=16)
// ---------------------------------------------------------------------------
static constexpr int Bsz = 2;
static constexpr int Sq  = 4096;
static constexpr int Dm  = 1024;           // model dim (= N = K of all GEMMs)
static constexpr int Mt  = Bsz * Sq;       // 8192 GEMM rows
static constexpr int Hh  = 16;
static constexpr int HD  = 64;             // head dim
static constexpr int WND = 16;             // window
static constexpr int LP  = 8;              // left pad (W/2)

typedef unsigned short u16;
typedef __attribute__((ext_vector_type(16))) __bf16 v16bf;
typedef __attribute__((ext_vector_type(8)))  float  v8f;

// ---------------------------------------------------------------------------
// bf16 helpers (round-to-nearest-even)
// ---------------------------------------------------------------------------
__device__ __forceinline__ u16 f2bf(float f) {
    unsigned int u = __float_as_uint(f);
    u += 0x7FFFu + ((u >> 16) & 1u);
    return (u16)(u >> 16);
}
__device__ __forceinline__ float bf2f(u16 h) {
    return __uint_as_float(((unsigned int)h) << 16);
}

// ---------------------------------------------------------------------------
// CDNA5 async global->LDS copy (ASYNCcnt-tracked, bypasses VGPRs).
// lds = wave-relative LDS byte address (low 32 bits of a flat shared pointer).
// ---------------------------------------------------------------------------
__device__ __forceinline__ void async_copy16(unsigned lds, const void* g) {
    asm volatile("global_load_async_to_lds_b128 %0, %1, off"
                 :: "v"(lds), "v"(g) : "memory");
}

// ---------------------------------------------------------------------------
// f32 -> bf16 conversion, 4 elements/thread (sizes are multiples of 1024)
// ---------------------------------------------------------------------------
__global__ __launch_bounds__(256) void cvt4_kernel(const float* __restrict__ in,
                                                   u16* __restrict__ out, int n4) {
    int i = blockIdx.x * blockDim.x + threadIdx.x;
    if (i >= n4) return;
    float4 f = reinterpret_cast<const float4*>(in)[i];
    uint2 p;
    p.x = (unsigned)f2bf(f.x) | ((unsigned)f2bf(f.y) << 16);
    p.y = (unsigned)f2bf(f.z) | ((unsigned)f2bf(f.w) << 16);
    reinterpret_cast<uint2*>(out)[i] = p;
}

// ---------------------------------------------------------------------------
// WMMA GEMM body:  C[M x N] = A[M x K] * B^T  (B stored row-major [N x K])
// Block tile 128x128, BK=32, 8 waves; wave tile 32x64 (2x4 of 16x16).
// Double-buffered LDS filled by async global->LDS copies; WMMA consumes the
// current buffer while the async engine streams the next K-tile.
// ---------------------------------------------------------------------------
template <bool BF16OUT>
__device__ __forceinline__ void gemm_body(const u16* __restrict__ A,
                                          const u16* __restrict__ Bw,
                                          u16* __restrict__ outb,
                                          float* __restrict__ outf) {
    __shared__ __align__(16) u16 As[2][128][40];   // 32 cols + pad
    __shared__ __align__(16) u16 Bs[2][128][40];

    const int tid = threadIdx.x;
    const int bm  = blockIdx.x * 128;
    const int bn  = blockIdx.y * 128;

    // cooperative tile fill: 2 threads per row, 16 bf16 (2 x async b128) each
    const int lrow = tid >> 1;
    const int lcol = (tid & 1) << 4;
    const u16* gA = A  + (long)(bm + lrow) * Dm + lcol;
    const u16* gB = Bw + (long)(bn + lrow) * Dm + lcol;

    unsigned ldsA[2], ldsB[2];
    ldsA[0] = (unsigned)(uintptr_t)&As[0][lrow][lcol];
    ldsA[1] = (unsigned)(uintptr_t)&As[1][lrow][lcol];
    ldsB[0] = (unsigned)(uintptr_t)&Bs[0][lrow][lcol];
    ldsB[1] = (unsigned)(uintptr_t)&Bs[1][lrow][lcol];

    // wave/lane geometry
    const int lane = tid & 31;
    const int lr   = lane & 15;
    const int hi   = lane >> 4;          // 0: lanes 0-15, 1: lanes 16-31
    const int wv   = tid >> 5;           // 8 waves
    const int wm   = (wv >> 1) * 32;     // wave M offset: 0,32,64,96
    const int wn   = (wv & 1) * 64;      // wave N offset: 0,64

    v8f acc[2][4];
#pragma unroll
    for (int mi = 0; mi < 2; ++mi)
#pragma unroll
        for (int ni = 0; ni < 4; ++ni)
            acc[mi][ni] = v8f{0.f, 0.f, 0.f, 0.f, 0.f, 0.f, 0.f, 0.f};

    auto fill = [&](int buf, int kk) {
        async_copy16(ldsA[buf],      gA + kk);       // 4 async ops / thread
        async_copy16(ldsA[buf] + 16, gA + kk + 8);
        async_copy16(ldsB[buf],      gB + kk);
        async_copy16(ldsB[buf] + 16, gB + kk + 8);
    };

    fill(0, 0);                                      // prologue

    for (int kk = 0, it = 0; kk < Dm; kk += 32, ++it) {
        const int cur = it & 1;
        if (kk + 32 < Dm) {
            fill(1 - cur, kk + 32);                  // overlap with compute
            if (kk + 64 < Dm) {                      // warm GL2 two tiles out
                __builtin_prefetch(gA + kk + 64, 0, 1);
                __builtin_prefetch(gB + kk + 64, 0, 1);
            }
            // async loads complete in order: <=4 outstanding means the
            // current tile's 4 copies have landed in LDS.
            asm volatile("s_wait_asynccnt 0x4" ::: "memory");
        } else {
            asm volatile("s_wait_asynccnt 0x0" ::: "memory");
        }
        __syncthreads();                             // all waves' fills visible

        union Frag { uint4 u[2]; v16bf v; };
        Frag fa[2], fb[4];
        // A 16x32 bf16: lane holds row (lane&15); K chunks {K0..K0+7, K0+16..K0+23},
        // K0 = (lane<16) ? 0 : 8
#pragma unroll
        for (int mi = 0; mi < 2; ++mi) {
            const u16* p = &As[cur][wm + mi * 16 + lr][hi * 8];
            fa[mi].u[0] = *reinterpret_cast<const uint4*>(p);
            fa[mi].u[1] = *reinterpret_cast<const uint4*>(p + 16);
        }
        // B 32x16 bf16: lane = column (lane&15); K chunks {K0..K0+7, K0+16..K0+23},
        // K0 = (lane<16) ? 0 : 8.  Column n of B == row n of W-tile.
#pragma unroll
        for (int ni = 0; ni < 4; ++ni) {
            const u16* p = &Bs[cur][wn + ni * 16 + lr][hi * 8];
            fb[ni].u[0] = *reinterpret_cast<const uint4*>(p);
            fb[ni].u[1] = *reinterpret_cast<const uint4*>(p + 16);
        }
#pragma unroll
        for (int mi = 0; mi < 2; ++mi)
#pragma unroll
            for (int ni = 0; ni < 4; ++ni)
                acc[mi][ni] = __builtin_amdgcn_wmma_f32_16x16x32_bf16(
                    false, fa[mi].v, false, fb[ni].v, (short)0, acc[mi][ni],
                    false, false);

        // frag ds_loads are dscnt-complete before the WMMAs above, so after
        // this barrier no wave can still be reading buf[cur] when it is
        // refilled next iteration.
        __syncthreads();
    }

    // C/D layout: VGPR r -> row r (lanes 0-15) / row r+8 (lanes 16-31); col = lane&15
#pragma unroll
    for (int mi = 0; mi < 2; ++mi) {
#pragma unroll
        for (int ni = 0; ni < 4; ++ni) {
            const int col     = bn + wn + ni * 16 + lr;
            const int rowbase = bm + wm + mi * 16 + hi * 8;
#pragma unroll
            for (int r = 0; r < 8; ++r) {
                const long idx = (long)(rowbase + r) * Dm + col;
                if (BF16OUT) outb[idx] = f2bf(acc[mi][ni][r]);
                else         outf[idx] = acc[mi][ni][r];
            }
        }
    }
}

__global__ __launch_bounds__(256) void gemm_qkv_kernel(
    const u16* __restrict__ x,
    const u16* __restrict__ wq, const u16* __restrict__ wk,
    const u16* __restrict__ wv,
    u16* __restrict__ q, u16* __restrict__ k, u16* __restrict__ v) {
    const u16* Bw = (blockIdx.z == 0) ? wq : (blockIdx.z == 1) ? wk : wv;
    u16* out      = (blockIdx.z == 0) ? q  : (blockIdx.z == 1) ? k  : v;
    gemm_body<true>(x, Bw, out, nullptr);
}

__global__ __launch_bounds__(256) void gemm_o_kernel(
    const u16* __restrict__ a, const u16* __restrict__ wo,
    float* __restrict__ out) {
    gemm_body<false>(a, wo, nullptr, out);
}

// ---------------------------------------------------------------------------
// Sliding-window attention: one block = (b, h, 128 query rows).
// k/v windows staged in LDS (bf16, padded stride), f32 softmax per thread.
// ---------------------------------------------------------------------------
__global__ __launch_bounds__(128) void attn_kernel(const u16* __restrict__ q,
                                                   const u16* __restrict__ k,
                                                   const u16* __restrict__ v,
                                                   u16* __restrict__ o) {
    const int b   = blockIdx.z;
    const int h   = blockIdx.y;
    const int s0  = blockIdx.x * 128;
    const int tid = threadIdx.x;

    // rows cover positions [s0-8, s0+134] -> 143 rows
    __shared__ __align__(16) u16 kb[143][72];
    __shared__ __align__(16) u16 vb[143][72];

    const long headBase = ((long)b * Sq) * Dm + (long)h * HD;

    for (int t = tid; t < 143 * 16; t += 128) {
        const int row = t >> 4;
        const int seg = (t & 15) << 2;            // 4 bf16 = 8 bytes
        const int pos = s0 - LP + row;
        uint2 kd = make_uint2(0u, 0u), vd = make_uint2(0u, 0u);
        if (pos >= 0 && pos < Sq) {
            const long g = headBase + (long)pos * Dm + seg;
            kd = *reinterpret_cast<const uint2*>(k + g);
            vd = *reinterpret_cast<const uint2*>(v + g);
        }
        *reinterpret_cast<uint2*>(&kb[row][seg]) = kd;
        *reinterpret_cast<uint2*>(&vb[row][seg]) = vd;
    }
    __syncthreads();

    const int  s  = s0 + tid;
    const long qg = headBase + (long)s * Dm;

    float qr[HD];
#pragma unroll
    for (int d = 0; d < HD; ++d) qr[d] = bf2f(q[qg + d]);

    float sc[WND];
    float mx = -INFINITY;
#pragma unroll
    for (int w = 0; w < WND; ++w) {
        const int pos = s - LP + w;
        const u16* kr = kb[tid + w];
        float dot = 0.f;
#pragma unroll
        for (int d = 0; d < HD; ++d) dot += qr[d] * bf2f(kr[d]);
        sc[w] = (pos >= 0 && pos < Sq) ? dot * 0.125f : -INFINITY; // 1/sqrt(64)
        mx = fmaxf(mx, sc[w]);
    }
    float sum = 0.f;
#pragma unroll
    for (int w = 0; w < WND; ++w) { sc[w] = __expf(sc[w] - mx); sum += sc[w]; }
    const float inv = 1.0f / sum;

    float acc[HD];
#pragma unroll
    for (int d = 0; d < HD; ++d) acc[d] = 0.f;
#pragma unroll
    for (int w = 0; w < WND; ++w) {
        const float pw = sc[w] * inv;
        const u16* vr = vb[tid + w];
#pragma unroll
        for (int d = 0; d < HD; ++d) acc[d] += pw * bf2f(vr[d]);
    }
#pragma unroll
    for (int d = 0; d < HD; ++d) o[qg + d] = f2bf(acc[d]);
}

// ---------------------------------------------------------------------------
// Launcher
// ---------------------------------------------------------------------------
extern "C" void kernel_launch(void* const* d_in, const int* in_sizes, int n_in,
                              void* d_out, int out_size, void* d_ws, size_t ws_size,
                              hipStream_t stream) {
    const float* x  = (const float*)d_in[0];
    const float* Wq = (const float*)d_in[1];
    const float* Wk = (const float*)d_in[2];
    const float* Wv = (const float*)d_in[3];
    const float* Wo = (const float*)d_in[4];
    float* out = (float*)d_out;

    const size_t XN = (size_t)Mt * Dm;    // 8,388,608
    const size_t WN = (size_t)Dm * Dm;    // 1,048,576

    u16* xb  = (u16*)d_ws;
    u16* wqb = xb  + XN;
    u16* wkb = wqb + WN;
    u16* wvb = wkb + WN;
    u16* wob = wvb + WN;
    u16* qb  = wob + WN;
    u16* kb  = qb  + XN;
    u16* vb  = kb  + XN;
    u16* ab  = vb  + XN;                  // total ~92.3 MB of workspace

    // 1) downconvert inputs to bf16
    cvt4_kernel<<<XN / 1024, 256, 0, stream>>>(x,  xb,  (int)(XN / 4));
    cvt4_kernel<<<WN / 1024, 256, 0, stream>>>(Wq, wqb, (int)(WN / 4));
    cvt4_kernel<<<WN / 1024, 256, 0, stream>>>(Wk, wkb, (int)(WN / 4));
    cvt4_kernel<<<WN / 1024, 256, 0, stream>>>(Wv, wvb, (int)(WN / 4));
    cvt4_kernel<<<WN / 1024, 256, 0, stream>>>(Wo, wob, (int)(WN / 4));

    // 2) fused Q/K/V projections (WMMA bf16, f32 accumulate, async LDS fills)
    gemm_qkv_kernel<<<dim3(Mt / 128, Dm / 128, 3), 256, 0, stream>>>(
        xb, wqb, wkb, wvb, qb, kb, vb);

    // 3) sliding-window attention
    attn_kernel<<<dim3(Sq / 128, Hh, Bsz), 128, 0, stream>>>(qb, kb, vb, ab);

    // 4) output projection (WMMA bf16 -> f32 out)
    gemm_o_kernel<<<dim3(Mt / 128, Dm / 128, 1), 256, 0, stream>>>(ab, wob, out);
}
